// QuantizedLinear_6906307412080
// MI455X (gfx1250) — matrix operand order repaired and checked by reading, not compiled, not run on hardware
//
#include <hip/hip_runtime.h>

typedef __attribute__((ext_vector_type(16))) _Float16 v16h;
typedef __attribute__((ext_vector_type(8)))  _Float16 v8h;
typedef __attribute__((ext_vector_type(4)))  _Float16 v4h;
typedef __attribute__((ext_vector_type(8)))  float    v8f;
typedef __attribute__((ext_vector_type(4)))  float    v4f;
typedef __attribute__((ext_vector_type(4)))  int      v4i;

#define TOKENS_ 8192
#define IN_F_   4096
#define OUT_F_  16384

#define BM 128
#define BN 128
#define BK 32
#define LDSW (BK + 8)   // 40 halfs = 80 B row stride: 16B-aligned, conflict-free

static __device__ __forceinline__ v16h join16(v8h lo, v8h hi) {
    union { v16h v; v8h h[2]; } u;
    u.h[0] = lo; u.h[1] = hi;
    return u.v;
}

__global__ __launch_bounds__(256)
void qlinear_wmma_kernel(const float* __restrict__ x,
                         const int*   __restrict__ wq,
                         const float* __restrict__ bias,
                         const float* __restrict__ scale_p,
                         const int*   __restrict__ zp_p,
                         float*       __restrict__ out)
{
    __shared__ _Float16 sA[BM * LDSW];   // x tile, f16, [m][k]
    __shared__ _Float16 sB[BN * LDSW];   // dequantized w tile, f16, [n][k]

    const int tid   = threadIdx.x;
    const int lane  = tid & 31;
    const int wid   = tid >> 5;
    const int waveN = wid & 1;     // 2 waves along N (64 cols each)
    const int waveM = wid >> 1;    // 4 waves along M (32 rows each)

    const int tileM0 = blockIdx.y * BM;
    const int tileN0 = blockIdx.x * BN;

    const float scale = scale_p[0];
    const float zp    = (float)zp_p[0];

    // Per-lane fragment addressing (ISA 7.12.2, wave32, 16-bit operands)
    const int fr  = lane & 15;          // row-in-tile: M for A-frag, N for B-frag
    const int hi  = (lane >> 4) & 1;    // lane half
    const int kaA = hi * 8;             // A: chunks at K = hi*8 and hi*8+16
    const int kaB = hi * 8;             // B: chunks at K = hi*8 and hi*8+16

    v8f acc[2][4];
    #pragma unroll
    for (int mt = 0; mt < 2; ++mt)
        #pragma unroll
        for (int nt = 0; nt < 4; ++nt)
            acc[mt][nt] = (v8f){0.f,0.f,0.f,0.f,0.f,0.f,0.f,0.f};

    for (int k0 = 0; k0 < IN_F_; k0 += BK) {
        // ---- stage 128x32 tiles: 256 threads x 4 chunks of 4 elements ----
        v4f aReg[4];
        v4i bReg[4];
        #pragma unroll
        for (int j = 0; j < 4; ++j) {
            const int c   = tid + j * 256;      // 0..1023 flat chunk id
            const int row = c >> 3;             // 0..127
            const int kp  = (c & 7) << 2;       // 0..28
            aReg[j] = *(const v4f*)(x + (size_t)(tileM0 + row) * IN_F_ + k0 + kp);
            // weights are streamed exactly once: non-temporal keeps x hot in L2
            bReg[j] = __builtin_nontemporal_load(
                          (const v4i*)(wq + (size_t)(tileN0 + row) * IN_F_ + k0 + kp));
        }

        __syncthreads();   // prior iteration's LDS reads complete

        #pragma unroll
        for (int j = 0; j < 4; ++j) {
            const int c   = tid + j * 256;
            const int row = c >> 3;
            const int kp  = (c & 7) << 2;
            v4h ha, hb;
            ha.x = (_Float16)aReg[j].x;  ha.y = (_Float16)aReg[j].y;
            ha.z = (_Float16)aReg[j].z;  ha.w = (_Float16)aReg[j].w;
            hb.x = (_Float16)((float)bReg[j].x - zp);
            hb.y = (_Float16)((float)bReg[j].y - zp);
            hb.z = (_Float16)((float)bReg[j].z - zp);
            hb.w = (_Float16)((float)bReg[j].w - zp);
            *(v4h*)&sA[row * LDSW + kp] = ha;
            *(v4h*)&sB[row * LDSW + kp] = hb;
        }

        __syncthreads();   // tiles visible to all waves

        // ---- fragment loads (ds_read_b128) + WMMA ----
        v16h aFrag[2], bFrag[4];
        #pragma unroll
        for (int mt = 0; mt < 2; ++mt) {
            const _Float16* pa = &sA[(waveM * 32 + mt * 16 + fr) * LDSW + kaA];
            aFrag[mt] = join16(*(const v8h*)pa, *(const v8h*)(pa + 16));
        }
        #pragma unroll
        for (int nt = 0; nt < 4; ++nt) {
            const _Float16* pb = &sB[(waveN * 64 + nt * 16 + fr) * LDSW + kaB];
            bFrag[nt] = join16(*(const v8h*)pb, *(const v8h*)(pb + 16));
        }
        #pragma unroll
        for (int mt = 0; mt < 2; ++mt)
            #pragma unroll
            for (int nt = 0; nt < 4; ++nt)
                acc[mt][nt] = __builtin_amdgcn_wmma_f32_16x16x32_f16(
                    /*neg_a=*/false, aFrag[mt],
                    /*neg_b=*/false, bFrag[nt],
                    /*c_mod=*/(short)0, acc[mt][nt],
                    /*reuse_a=*/false, /*reuse_b=*/false);
    }

    // ---- epilogue: D layout = lane 0-15 -> M=r, lane 16-31 -> M=r+8; N=lane&15
    const int colLo = lane & 15;
    const int rowHi = hi * 8;
    #pragma unroll
    for (int nt = 0; nt < 4; ++nt) {
        const int col = tileN0 + waveN * 64 + nt * 16 + colLo;
        const float b = bias[col];
        #pragma unroll
        for (int mt = 0; mt < 2; ++mt) {
            const int rBase = tileM0 + waveM * 32 + mt * 16 + rowHi;
            #pragma unroll
            for (int r = 0; r < 8; ++r) {
                const float v = acc[mt][nt][r] * scale + b;
                // output is written once and never re-read: stream past L2
                __builtin_nontemporal_store(v, out + (size_t)(rBase + r) * OUT_F_ + col);
            }
        }
    }
}

extern "C" void kernel_launch(void* const* d_in, const int* in_sizes, int n_in,
                              void* d_out, int out_size, void* d_ws, size_t ws_size,
                              hipStream_t stream) {
    (void)in_sizes; (void)n_in; (void)d_ws; (void)ws_size; (void)out_size;
    const float* x     = (const float*)d_in[0];
    const int*   wq    = (const int*)  d_in[1];
    const float* bias  = (const float*)d_in[2];
    const float* scale = (const float*)d_in[3];
    const int*   zp    = (const int*)  d_in[4];
    float*       out   = (float*)d_out;

    dim3 grid(OUT_F_ / BN, TOKENS_ / BM);   // (128, 64)
    dim3 block(256);                        // 8 wave32s
    qlinear_wmma_kernel<<<grid, block, 0, stream>>>(x, wq, bias, scale, zp, out);
}
